// SFAU_5076651343985
// MI455X (gfx1250) — matrix operand order repaired and checked by reading, not compiled, not run on hardware
//
#include <hip/hip_runtime.h>
#include <hip/hip_bf16.h>
#include <math.h>

// ---- problem constants (match reference shapes) ----
#define B_   2
#define C_   64
#define H_   128
#define W_   128
#define S_   4
#define SH_  (H_*S_)       // 512
#define SW_  (W_*S_)       // 512
#define HW_  (H_*W_)       // 16384
#define SHW_ (SH_*SW_)     // 262144
#define KU_  5
#define E_   64

typedef __bf16 bf16_t;
typedef __attribute__((ext_vector_type(16))) __bf16 v16bf;
typedef __attribute__((ext_vector_type(8)))  float  v8f;

// A-matrix (16x32 bf16) per-lane element e -> K index (ISA 7.12.2)
__device__ __forceinline__ int a_kidx(int e, int lane) {
    int v = e >> 1, h = e & 1;
    return ((lane & 16) ? 8 : 0) + ((v & 3) * 2 + h) + ((v >> 2) << 4);
}

// ---- CDNA5 async global->LDS copy (ASYNCcnt path, ISA ch.10 / 08_async) ----
__device__ __forceinline__ void async_g2l_b32(void* lds_ptr, const float* gptr) {
    unsigned lds_off = (unsigned)(unsigned long long)lds_ptr; // LDS aperture: low 32 bits = LDS addr
    asm volatile("global_load_async_to_lds_b32 %0, %1, off"
                 :: "v"(lds_off), "v"(gptr)
                 : "memory");
}
__device__ __forceinline__ void wait_async0() {
    asm volatile("s_wait_asynccnt 0x0" ::: "memory");
}

// ---------------- bicubic upsample (Keys a=-0.5, half-pixel) ----------------
__device__ __forceinline__ void cubw(float f, float w[4]) {
    const float A = -0.5f;
    float t;
    t = 1.f + f; w[0] = ((A * t - 5.f * A) * t + 8.f * A) * t - 4.f * A;
    t = f;       w[1] = ((A + 2.f) * t - (A + 3.f)) * t * t + 1.f;
    t = 1.f - f; w[2] = ((A + 2.f) * t - (A + 3.f)) * t * t + 1.f;
    t = 2.f - f; w[3] = ((A * t - 5.f * A) * t + 8.f * A) * t - 4.f * A;
}

// writes NCHW f32 (for pc1a) and pixel-major bf16 (for the gate WMMA GEMM)
__global__ void k_upsample(const float* __restrict__ x, float* __restrict__ up,
                           bf16_t* __restrict__ upb) {
    int idx = blockIdx.x * blockDim.x + threadIdx.x;
    if (idx >= B_ * C_ * SHW_) return;
    int ow = idx % SW_; int t = idx / SW_;
    int oh = t % SH_;  t /= SH_;
    int c  = t % C_;   int b = t / C_;
    float ih = (oh + 0.5f) * 0.25f - 0.5f;
    float iw = (ow + 0.5f) * 0.25f - 0.5f;
    int ih0 = (int)floorf(ih), iw0 = (int)floorf(iw);
    float wh[4], wv[4];
    cubw(ih - (float)ih0, wh);
    cubw(iw - (float)iw0, wv);
    const float* src = x + ((long)(b * C_ + c)) * HW_;
    float acc = 0.f;
    for (int ty = 0; ty < 4; ++ty) {
        int yy = ih0 - 1 + ty; yy = yy < 0 ? 0 : (yy > H_ - 1 ? H_ - 1 : yy);
        for (int tx = 0; tx < 4; ++tx) {
            int xx = iw0 - 1 + tx; xx = xx < 0 ? 0 : (xx > W_ - 1 ? W_ - 1 : xx);
            acc += wh[ty] * wv[tx] * src[yy * W_ + xx];
        }
    }
    up[idx] = acc;
    upb[((long)b * SHW_ + (long)oh * SW_ + ow) * 64 + c] = (bf16_t)acc;
}

// ---------------- pack (O x I) f32 weight -> bf16 B-layout [K=I][N=O] -------
__global__ void k_packw(const float* __restrict__ w, bf16_t* __restrict__ wb, int O, int I) {
    int i = blockIdx.x * blockDim.x + threadIdx.x;
    if (i >= O * I) return;
    int n = i / I, k = i % I;
    wb[k * O + n] = (bf16_t)w[n * I + k];
}

// ---------------- gate: 1x1 conv 64->64 + sigmoid, WMMA on bf16 rows --------
__global__ void k_gate(const bf16_t* __restrict__ A, const bf16_t* __restrict__ wb,
                       const float* __restrict__ bias, float* __restrict__ gate) {
    int tile = blockIdx.x, lane = threadIdx.x;
    long m0 = (long)tile * 16;
    int  b  = (int)(m0 / SHW_);
    long p0 = m0 % SHW_;
    int  mloc = lane & 15;
    const bf16_t* arow = A + (m0 + mloc) * 64;
    v16bf a[2];
    for (int ks = 0; ks < 2; ++ks)
        for (int e = 0; e < 16; ++e)
            a[ks][e] = arow[a_kidx(e, lane) + ks * 32];
    for (int nt = 0; nt < 4; ++nt) {
        int n = nt * 16 + (lane & 15);
        v8f acc = {};
        for (int ks = 0; ks < 2; ++ks) {
            v16bf bf_;
            int kb0 = ks * 32;
            for (int e = 0; e < 16; ++e) bf_[e] = wb[(kb0 + a_kidx(e, lane)) * 64 + n];
            acc = __builtin_amdgcn_wmma_f32_16x16x32_bf16(false, a[ks], false, bf_,
                                                          (short)0, acc, false, false);
        }
        float bv = bias[n];
        for (int r = 0; r < 8; ++r) {
            int m = r + ((lane >> 4) << 3);
            float v = acc[r] + bv;
            gate[((long)b * C_ + n) * SHW_ + p0 + m] = 1.f / (1.f + __expf(-v));
        }
    }
}

// ---------------- pc1a: 1x1 conv concat(y[3], up[64]) -> 16 -----------------
__global__ void k_pc1a(const float* __restrict__ y, const float* __restrict__ up,
                       const float* __restrict__ w, const float* __restrict__ bias,
                       float* __restrict__ out) {
    int idx = blockIdx.x * blockDim.x + threadIdx.x;
    if (idx >= B_ * SHW_) return;
    int p = idx % SHW_, b = idx / SHW_;
    float acc[16];
    for (int o = 0; o < 16; ++o) acc[o] = bias[o];
    for (int c = 0; c < 3; ++c) {
        float v = y[((long)(b * 3 + c)) * SHW_ + p];
        for (int o = 0; o < 16; ++o) acc[o] += w[o * 67 + c] * v;
    }
    for (int c = 0; c < 64; ++c) {
        float v = up[((long)(b * 64 + c)) * SHW_ + p];
        for (int o = 0; o < 16; ++o) acc[o] += w[o * 67 + 3 + c] * v;
    }
    for (int o = 0; o < 16; ++o) out[((long)(b * 16 + o)) * SHW_ + p] = acc[o];
}

// ---------------- conv 5x5 16->16, pad 2; weights staged via async->LDS -----
__global__ void k_conv5(const float* __restrict__ in, const float* __restrict__ w,
                        const float* __restrict__ bias, float* __restrict__ out) {
    __shared__ float wsh[16 * 16 * 25]; // 25.6 KB
    for (int i = threadIdx.x; i < 16 * 16 * 25; i += 128)
        async_g2l_b32(&wsh[i], w + i);
    wait_async0();
    __syncthreads();

    int idx = blockIdx.x * blockDim.x + threadIdx.x;
    int p = idx % SHW_, b = idx / SHW_;
    int ph = p / SW_, pw = p % SW_;
    float acc[16];
    for (int o = 0; o < 16; ++o) acc[o] = bias[o];
    for (int ci = 0; ci < 16; ++ci) {
        const float* src = in + ((long)(b * 16 + ci)) * SHW_;
        for (int dy = 0; dy < 5; ++dy) {
            int hh = ph + dy - 2;
            if (hh < 0 || hh >= SH_) continue;
            for (int dx = 0; dx < 5; ++dx) {
                int ww = pw + dx - 2;
                if (ww < 0 || ww >= SW_) continue;
                float v = src[hh * SW_ + ww];
                for (int o = 0; o < 16; ++o)
                    acc[o] += wsh[(o * 16 + ci) * 25 + dy * 5 + dx] * v;
            }
        }
    }
    for (int o = 0; o < 16; ++o) out[((long)(b * 16 + o)) * SHW_ + p] = acc[o];
}

// ---------------- conv 3x3 16->16, pad 1, optional relu / residual ----------
__global__ void k_conv3(const float* __restrict__ in, const float* __restrict__ w,
                        const float* __restrict__ bias, float* __restrict__ out,
                        int dorelu, const float* __restrict__ res) {
    __shared__ float wsh[16 * 16 * 9]; // 9.2 KB
    for (int i = threadIdx.x; i < 16 * 16 * 9; i += 128)
        async_g2l_b32(&wsh[i], w + i);
    wait_async0();
    __syncthreads();

    int idx = blockIdx.x * blockDim.x + threadIdx.x;
    int p = idx % SHW_, b = idx / SHW_;
    int ph = p / SW_, pw = p % SW_;
    float acc[16];
    for (int o = 0; o < 16; ++o) acc[o] = bias[o];
    for (int ci = 0; ci < 16; ++ci) {
        const float* src = in + ((long)(b * 16 + ci)) * SHW_;
        for (int dy = 0; dy < 3; ++dy) {
            int hh = ph + dy - 1;
            if (hh < 0 || hh >= SH_) continue;
            for (int dx = 0; dx < 3; ++dx) {
                int ww = pw + dx - 1;
                if (ww < 0 || ww >= SW_) continue;
                float v = src[hh * SW_ + ww];
                for (int o = 0; o < 16; ++o)
                    acc[o] += wsh[(o * 16 + ci) * 9 + dy * 3 + dx] * v;
            }
        }
    }
    for (int o = 0; o < 16; ++o) {
        float a = acc[o];
        if (dorelu) a = fmaxf(a, 0.f);
        if (res) a += res[((long)(b * 16 + o)) * SHW_ + p];
        out[((long)(b * 16 + o)) * SHW_ + p] = a;
    }
}

// ---------------- frequency branch: 1x1 -> relu -> 1x1 (16ch) ---------------
__global__ void k_frq(const float* __restrict__ in,
                      const float* __restrict__ w1, const float* __restrict__ b1,
                      const float* __restrict__ w2, const float* __restrict__ b2,
                      float* __restrict__ out) {
    int idx = blockIdx.x * blockDim.x + threadIdx.x;
    if (idx >= B_ * SHW_) return;
    int p = idx % SHW_, b = idx / SHW_;
    float v[16], t[16];
    for (int c = 0; c < 16; ++c) v[c] = in[((long)(b * 16 + c)) * SHW_ + p];
    for (int o = 0; o < 16; ++o) {
        float a = b1[o];
        for (int c = 0; c < 16; ++c) a += w1[o * 16 + c] * v[c];
        t[o] = fmaxf(a, 0.f);
    }
    for (int o = 0; o < 16; ++o) {
        float a = b2[o];
        for (int c = 0; c < 16; ++c) a += w2[o * 16 + c] * t[c];
        out[((long)(b * 16 + o)) * SHW_ + p] = a;
    }
}

// ---------------- cat: 1x1 conv concat(spa16, frq16) -> 8 -------------------
__global__ void k_cat(const float* __restrict__ spa, const float* __restrict__ frq,
                      const float* __restrict__ w, const float* __restrict__ bias,
                      float* __restrict__ f8) {
    int idx = blockIdx.x * blockDim.x + threadIdx.x;
    if (idx >= B_ * SHW_) return;
    int p = idx % SHW_, b = idx / SHW_;
    float v[32];
    for (int c = 0; c < 16; ++c) {
        v[c]      = spa[((long)(b * 16 + c)) * SHW_ + p];
        v[16 + c] = frq[((long)(b * 16 + c)) * SHW_ + p];
    }
    for (int o = 0; o < 8; ++o) {
        float a = bias[o];
        for (int c = 0; c < 32; ++c) a += w[o * 32 + c] * v[c];
        f8[((long)(b * 8 + o)) * SHW_ + p] = a;
    }
}

// ---------------- SE pool (mean over spatial) -------------------------------
__global__ void k_pool(const float* __restrict__ f8, float* __restrict__ pooled) {
    int bc = blockIdx.x; // 0..15 = b*8+c
    __shared__ float sm[256];
    float s = 0.f;
    for (int i = threadIdx.x; i < SHW_; i += 256) s += f8[(long)bc * SHW_ + i];
    sm[threadIdx.x] = s;
    __syncthreads();
    for (int step = 128; step > 0; step >>= 1) {
        if (threadIdx.x < step) sm[threadIdx.x] += sm[threadIdx.x + step];
        __syncthreads();
    }
    if (threadIdx.x == 0) pooled[bc] = sm[0] * (1.f / (float)SHW_);
}

// ---------------- SE MLP: 8 -> 2 relu -> 8 sigmoid --------------------------
__global__ void k_se(const float* __restrict__ pooled,
                     const float* __restrict__ w1, const float* __restrict__ b1,
                     const float* __restrict__ w2, const float* __restrict__ b2,
                     float* __restrict__ sew) {
    int i = threadIdx.x;
    if (i >= B_ * 8) return;
    int b = i >> 3, o = i & 7;
    float t0 = b1[0], t1 = b1[1];
    for (int c = 0; c < 8; ++c) {
        float v = pooled[b * 8 + c];
        t0 += w1[c] * v;
        t1 += w1[8 + c] * v;
    }
    t0 = fmaxf(t0, 0.f); t1 = fmaxf(t1, 0.f);
    float a = b2[o] + w2[o * 2 + 0] * t0 + w2[o * 2 + 1] * t1;
    sew[i] = 1.f / (1.f + __expf(-a));
}

// ---------------- pc3: (f8 * se) 1x1 conv 8 -> 64 = enc ---------------------
__global__ void k_pc3(const float* __restrict__ f8, const float* __restrict__ sew,
                      const float* __restrict__ w, const float* __restrict__ bias,
                      float* __restrict__ enc) {
    int idx = blockIdx.x * blockDim.x + threadIdx.x;
    if (idx >= B_ * SHW_) return;
    int p = idx % SHW_, b = idx / SHW_;
    float v[8];
    for (int c = 0; c < 8; ++c)
        v[c] = f8[((long)(b * 8 + c)) * SHW_ + p] * sew[b * 8 + c];
    for (int o = 0; o < 64; ++o) {
        float a = bias[o];
        for (int c = 0; c < 8; ++c) a += w[o * 8 + c] * v[c];
        enc[((long)(b * 64 + o)) * SHW_ + p] = a;
    }
}

// ---------------- layernorm over 64 channels (NCHW in) -> bf16 pixel-major --
__global__ void k_ln(const float* __restrict__ in, const float* __restrict__ g,
                     const float* __restrict__ be, bf16_t* __restrict__ out,
                     int nPix, int nTot) {
    int idx = blockIdx.x * blockDim.x + threadIdx.x;
    if (idx >= nTot) return;
    int p = idx % nPix, b = idx / nPix;
    const float* src = in + (long)b * 64 * (long)nPix + p;
    float m = 0.f;
    for (int c = 0; c < 64; ++c) m += src[(long)c * nPix];
    m *= (1.f / 64.f);
    float vv = 0.f;
    for (int c = 0; c < 64; ++c) { float d = src[(long)c * nPix] - m; vv += d * d; }
    vv *= (1.f / 64.f);
    float inv = rsqrtf(vv + 1e-5f);
    bf16_t* dst = out + (long)idx * 64;
    for (int c = 0; c < 64; ++c)
        dst[c] = (bf16_t)((src[(long)c * nPix] - m) * inv * g[c] + be[c]);
}

// ---------------- linear 64->64 on bf16 pixel-major rows, via WMMA ----------
__global__ void k_lin64(const bf16_t* __restrict__ A, const bf16_t* __restrict__ wb,
                        const float* __restrict__ bias, bf16_t* __restrict__ out) {
    int tile = blockIdx.x, lane = threadIdx.x;
    long m0 = (long)tile * 16;
    int  mloc = lane & 15;
    const bf16_t* arow = A + (m0 + mloc) * 64;
    v16bf a[2];
    for (int ks = 0; ks < 2; ++ks)
        for (int e = 0; e < 16; ++e)
            a[ks][e] = arow[a_kidx(e, lane) + ks * 32];
    for (int nt = 0; nt < 4; ++nt) {
        int n = nt * 16 + (lane & 15);
        v8f acc = {};
        for (int ks = 0; ks < 2; ++ks) {
            v16bf bf_;
            int kb0 = ks * 32;
            for (int e = 0; e < 16; ++e) bf_[e] = wb[(kb0 + a_kidx(e, lane)) * 64 + n];
            acc = __builtin_amdgcn_wmma_f32_16x16x32_bf16(false, a[ks], false, bf_,
                                                          (short)0, acc, false, false);
        }
        float bv = bias[n];
        for (int r = 0; r < 8; ++r) {
            int m = r + ((lane >> 4) << 3);
            out[(m0 + m) * 64 + n] = (bf16_t)(acc[r] + bv);
        }
    }
}

// ---------------- attention: sim (WMMA) + softmax + atn (WMMA) + blend ------
// one wave per low-res cell; 16 queries (its 4x4 block) share 25 keys/values
__global__ void k_attn(const bf16_t* __restrict__ q, const bf16_t* __restrict__ kb,
                       const float* __restrict__ x, const float* __restrict__ gate,
                       const float* __restrict__ enc,
                       float* __restrict__ outF, float* __restrict__ outS) {
    int cell = blockIdx.x, lane = threadIdx.x;
    int j = cell % W_; int t = cell / W_;
    int i = t % H_;    int b = t / H_;
    int mloc = lane & 15;
    int sh = i * 4 + (mloc >> 2), sw = j * 4 + (mloc & 3);

    // A = q tile (16 x 64)
    const bf16_t* qrow = q + (((long)b * SH_ + sh) * SW_ + sw) * 64;
    v16bf aq[2];
    for (int ks = 0; ks < 2; ++ks)
        for (int e = 0; e < 16; ++e)
            aq[ks][e] = qrow[a_kidx(e, lane) + ks * 32];

    // sim = q @ K^T over 25 patches (padded to 32)
    float sv[2][8];
    for (int nt = 0; nt < 2; ++nt) {
        int p = nt * 16 + (lane & 15);
        int di = p / 5 - 2, dj = p % 5 - 2;
        int hh = i + di, ww = j + dj;
        bool ok = (p < 25) && hh >= 0 && hh < H_ && ww >= 0 && ww < W_;
        const bf16_t* krow = kb + (((long)b * H_ + (ok ? hh : 0)) * W_ + (ok ? ww : 0)) * 64;
        v8f acc = {};
        for (int ks = 0; ks < 2; ++ks) {
            v16bf bf_;
            int kb0 = ks * 32;
            for (int e = 0; e < 16; ++e) bf_[e] = ok ? krow[kb0 + a_kidx(e, lane)] : (bf16_t)0.f;
            acc = __builtin_amdgcn_wmma_f32_16x16x32_bf16(false, aq[ks], false, bf_,
                                                          (short)0, acc, false, false);
        }
        for (int r = 0; r < 8; ++r) sv[nt][r] = (p < 25) ? acc[r] : -1e30f;
    }

    // row-wise softmax over 25 entries: reduce over lanes within each 16-group
    float av[2][8];
    for (int r = 0; r < 8; ++r) {
        float mx = fmaxf(sv[0][r], sv[1][r]);
        for (int msk = 1; msk < 16; msk <<= 1) mx = fmaxf(mx, __shfl_xor(mx, msk, 32));
        float e0 = __expf(sv[0][r] - mx);
        float e1 = __expf(sv[1][r] - mx);
        float sum = e0 + e1;
        for (int msk = 1; msk < 16; msk <<= 1) sum += __shfl_xor(sum, msk, 32);
        float inv = 1.f / sum;
        av[0][r] = e0 * inv;
        av[1][r] = e1 * inv;
    }

    // re-fragment attn (16x32) through LDS into WMMA A-layout
    __shared__ float lat[16 * 32];
    for (int nt = 0; nt < 2; ++nt)
        for (int r = 0; r < 8; ++r) {
            int M = r + ((lane >> 4) << 3);
            int N = (lane & 15) + nt * 16;
            lat[M * 32 + N] = av[nt][r];
        }
    __syncthreads();
    v16bf aat;
    for (int e = 0; e < 16; ++e)
        aat[e] = (bf16_t)lat[mloc * 32 + a_kidx(e, lane)];

    // out = attn @ V (V = x patches, 32 x 64), fused gate blend on store
    for (int nt = 0; nt < 4; ++nt) {
        int n = nt * 16 + (lane & 15); // channel
        v16bf bf_;
        int kb0 = 0;
        for (int e = 0; e < 16; ++e) {
            int p = kb0 + a_kidx(e, lane);
            int di = p / 5 - 2, dj = p % 5 - 2;
            int hh = i + di, ww = j + dj;
            bool ok = (p < 25) && hh >= 0 && hh < H_ && ww >= 0 && ww < W_;
            bf_[e] = ok ? (bf16_t)x[((long)b * 64 + n) * HW_ + hh * W_ + ww] : (bf16_t)0.f;
        }
        v8f acc = {};
        acc = __builtin_amdgcn_wmma_f32_16x16x32_bf16(false, aat, false, bf_,
                                                      (short)0, acc, false, false);
        for (int r = 0; r < 8; ++r) {
            int M = r + ((lane >> 4) << 3);
            int sh2 = i * 4 + (M >> 2), sw2 = j * 4 + (M & 3);
            long off = ((long)b * 64 + n) * SHW_ + (long)sh2 * SW_ + sw2;
            float sval = acc[r];
            float g = gate[off];
            outS[off] = sval;
            outF[off] = g * sval + (1.f - g) * enc[off];
        }
    }
}

// ---------------------------------------------------------------------------
extern "C" void kernel_launch(void* const* d_in, const int* in_sizes, int n_in,
                              void* d_out, int out_size, void* d_ws, size_t ws_size,
                              hipStream_t stream) {
    (void)in_sizes; (void)n_in; (void)out_size; (void)ws_size;
    const float* x      = (const float*)d_in[0];
    const float* y      = (const float*)d_in[1];
    const float* gate_w = (const float*)d_in[2];
    const float* gate_b = (const float*)d_in[3];
    const float* pc1a_w = (const float*)d_in[4];
    const float* pc1a_b = (const float*)d_in[5];
    const float* pc1b_w = (const float*)d_in[6];
    const float* pc1b_b = (const float*)d_in[7];
    const float* spa1_w = (const float*)d_in[8];
    const float* spa1_b = (const float*)d_in[9];
    const float* spa2_w = (const float*)d_in[10];
    const float* spa2_b = (const float*)d_in[11];
    const float* frq1_w = (const float*)d_in[12];
    const float* frq1_b = (const float*)d_in[13];
    const float* frq2_w = (const float*)d_in[14];
    const float* frq2_b = (const float*)d_in[15];
    const float* cat_w  = (const float*)d_in[16];
    const float* cat_b  = (const float*)d_in[17];
    const float* ca1_w  = (const float*)d_in[18];
    const float* ca1_b  = (const float*)d_in[19];
    const float* ca2_w  = (const float*)d_in[20];
    const float* ca2_b  = (const float*)d_in[21];
    const float* pc3_w  = (const float*)d_in[22];
    const float* pc3_b  = (const float*)d_in[23];
    const float* lny_g  = (const float*)d_in[24];
    const float* lny_b  = (const float*)d_in[25];
    const float* lnx_g  = (const float*)d_in[26];
    const float* lnx_b  = (const float*)d_in[27];
    const float* q_w    = (const float*)d_in[28];
    const float* q_b    = (const float*)d_in[29];
    const float* k_w    = (const float*)d_in[30];
    const float* k_b    = (const float*)d_in[31];

    const size_t OUTN = (size_t)B_ * C_ * SHW_;
    float* out_final = (float*)d_out;
    float* out_sapa  = out_final + OUTN;
    float* out_enc   = out_sapa + OUTN;

    char* wsp = (char*)d_ws;
    auto alloc = [&](size_t bytes) -> void* {
        void* p = (void*)wsp;
        wsp += (bytes + 255) & ~(size_t)255;
        return p;
    };
    float*  up    = (float*)alloc(OUTN * 4);
    bf16_t* upb   = (bf16_t*)alloc((size_t)B_ * SHW_ * 64 * 2);
    float*  gate  = (float*)alloc(OUTN * 4);
    float*  fbuf  = (float*)alloc((size_t)B_ * 16 * SHW_ * 4);
    float*  tbuf  = (float*)alloc((size_t)B_ * 16 * SHW_ * 4);
    float*  spa   = (float*)alloc((size_t)B_ * 16 * SHW_ * 4);
    float*  pooled= (float*)alloc(64 * 4);
    float*  sew   = (float*)alloc(64 * 4);
    bf16_t* wg_b  = (bf16_t*)alloc(4096 * 2);
    bf16_t* wq_b  = (bf16_t*)alloc(4096 * 2);
    bf16_t* wk_b  = (bf16_t*)alloc(4096 * 2);
    bf16_t* yn    = (bf16_t*)alloc((size_t)B_ * SHW_ * 64 * 2);
    bf16_t* qb    = (bf16_t*)alloc((size_t)B_ * SHW_ * 64 * 2);
    bf16_t* xn    = (bf16_t*)alloc((size_t)B_ * HW_ * 64 * 2);
    bf16_t* kbuf  = (bf16_t*)alloc((size_t)B_ * HW_ * 64 * 2);

    const int PIX = B_ * SHW_; // 524288 hi-res pixels

    k_upsample<<<(B_ * C_ * SHW_ + 255) / 256, 256, 0, stream>>>(x, up, upb);
    k_packw<<<16, 256, 0, stream>>>(gate_w, wg_b, 64, 64);
    k_packw<<<16, 256, 0, stream>>>(q_w, wq_b, 64, 64);
    k_packw<<<16, 256, 0, stream>>>(k_w, wk_b, 64, 64);

    k_gate<<<PIX / 16, 32, 0, stream>>>(upb, wg_b, gate_b, gate);

    k_pc1a<<<PIX / 256, 256, 0, stream>>>(y, up, pc1a_w, pc1a_b, tbuf);
    k_conv5<<<PIX / 128, 128, 0, stream>>>(tbuf, pc1b_w, pc1b_b, fbuf);
    k_conv3<<<PIX / 128, 128, 0, stream>>>(fbuf, spa1_w, spa1_b, tbuf, 1, nullptr);
    k_conv3<<<PIX / 128, 128, 0, stream>>>(tbuf, spa2_w, spa2_b, spa, 0, fbuf);
    k_frq<<<PIX / 256, 256, 0, stream>>>(fbuf, frq1_w, frq1_b, frq2_w, frq2_b, tbuf);
    k_cat<<<PIX / 256, 256, 0, stream>>>(spa, tbuf, cat_w, cat_b, fbuf); // f8 in fbuf

    k_pool<<<B_ * 8, 256, 0, stream>>>(fbuf, pooled);
    k_se<<<1, 32, 0, stream>>>(pooled, ca1_w, ca1_b, ca2_w, ca2_b, sew);
    k_pc3<<<PIX / 256, 256, 0, stream>>>(fbuf, sew, pc3_w, pc3_b, out_enc);

    k_ln<<<PIX / 256, 256, 0, stream>>>(out_enc, lny_g, lny_b, yn, SHW_, PIX);
    k_ln<<<(B_ * HW_) / 256, 256, 0, stream>>>(x, lnx_g, lnx_b, xn, HW_, B_ * HW_);
    k_lin64<<<PIX / 16, 32, 0, stream>>>(yn, wq_b, q_b, qb);
    k_lin64<<<(B_ * HW_) / 16, 32, 0, stream>>>(xn, wk_b, k_b, kbuf);

    k_attn<<<B_ * H_ * W_, 32, 0, stream>>>(qb, kbuf, x, gate, out_enc,
                                            out_final, out_sapa);
}
